// WindowCrossAttention3D_16552803959355
// MI455X (gfx1250) — matrix operand order repaired and checked by reading, not compiled, not run on hardware
//
#include <hip/hip_runtime.h>

// ---------------------------------------------------------------- types ----
typedef _Float16 half_t;
typedef __attribute__((ext_vector_type(16))) _Float16 v16h;
typedef __attribute__((ext_vector_type(8)))  _Float16 h8;
typedef __attribute__((ext_vector_type(8)))  float    v8f;

union V16 { v16h v; h8 h[2]; };

#define DEV __device__ __forceinline__

// Problem constants
static constexpr int Bc = 128;
static constexpr int Nc = 512;
static constexpr int Cc = 192;
static constexpr int Hc = 6;
static constexpr int Dc = 32;
static constexpr int Mrows = Bc * Nc;          // 65536
static constexpr float LOGIT_MAX = 4.605170185988091f; // ln(100)

// ------------------------------------------------------ fragment loaders ----
// A-fragment (16x32 f16, MxK): lane l<16 -> row l, K {0..7,16..23};
// lane l>=16 -> row l-16, K {8..15,24..31}.  base points at [row][k].
DEV v16h load_afrag(const half_t* base, int stride, int row0, int lane) {
  const int r = row0 + (lane & 15);
  const half_t* p = base + r * stride + ((lane & 16) ? 8 : 0);
  V16 t;
  t.h[0] = *(const h8*)(p);
  t.h[1] = *(const h8*)(p + 16);
  return t.v;
}

// B-fragment (32x16 f16, KxN) from row-major [n][k] storage:
// lane l<16 -> col n0+l, K {0..7,16..23}; lane l>=16 -> col n0+l-16, K {8..15,24..31}.
DEV v16h load_bfrag(const half_t* base, int stride, int n0, int lane) {
  const int n = n0 + (lane & 15);
  const half_t* p = base + n * stride + ((lane & 16) ? 8 : 0);
  V16 t;
  t.h[0] = *(const h8*)(p);
  t.h[1] = *(const h8*)(p + 16);
  return t.v;
}

// XOR-butterfly via ds_swizzle_b32 (group-of-32 mode: offset = xor<<10 | and=0x1f).
// Masks <= 8 keep the exchange inside each 16-lane half (matching C-frag rows).
template <int XM>
DEV float swz(float v) {
  int r = __builtin_amdgcn_ds_swizzle(__float_as_int(v), (XM << 10) | 0x1f);
  return __int_as_float(r);
}
DEV float redmax16(float v) {
  v = fmaxf(v, swz<1>(v)); v = fmaxf(v, swz<2>(v));
  v = fmaxf(v, swz<4>(v)); v = fmaxf(v, swz<8>(v));
  return v;
}
DEV float redsum16(float v) {
  v += swz<1>(v); v += swz<2>(v); v += swz<4>(v); v += swz<8>(v);
  return v;
}

// ------------------------------------------------------- small utilities ----
__global__ void f2h_kernel(const float* __restrict__ s, half_t* __restrict__ d, int n) {
  int i = blockIdx.x * blockDim.x + threadIdx.x;
  if (i < n) d[i] = (half_t)s[i];
}

__global__ void build_kvbias(const float* __restrict__ v_b, float* __restrict__ kvb) {
  int i = blockIdx.x * blockDim.x + threadIdx.x;
  if (i < 2 * Cc) kvb[i] = (i < Cc) ? 0.0f : v_b[i - Cc];
}

// In-place L2 normalization of [rows][32] fp16; optional per-head logit scale
// folded into Q. rows are ordered [B,H,N] so head = (row/512)%6.
__global__ void l2norm_rows(half_t* __restrict__ buf,
                            const float* __restrict__ logit_scale, int nrows) {
  int row = blockIdx.x * blockDim.x + threadIdx.x;
  if (row >= nrows) return;
  half_t* p = buf + (size_t)row * Dc;
  float vals[Dc];
  float ss = 0.0f;
  #pragma unroll
  for (int j = 0; j < Dc / 8; ++j) {
    h8 v = *(const h8*)(p + j * 8);
    #pragma unroll
    for (int e = 0; e < 8; ++e) { float f = (float)v[e]; vals[j * 8 + e] = f; ss += f * f; }
  }
  float scale = 1.0f;
  if (logit_scale) {
    int h = (row >> 9) % Hc;
    scale = __expf(fminf(logit_scale[h], LOGIT_MAX));
  }
  float inv = scale / fmaxf(sqrtf(ss), 1e-12f);
  #pragma unroll
  for (int j = 0; j < Dc / 8; ++j) {
    h8 v;
    #pragma unroll
    for (int e = 0; e < 8; ++e) v[e] = (half_t)(vals[j * 8 + e] * inv);
    *(h8*)(p + j * 8) = v;
  }
}

// ------------------------------------------------------------ WMMA GEMM ----
// C[M,Ncol] = A[M,K] * W[Ncol,K]^T + bias
// MODE 0: fp32 row-major output (out0)
// MODE 1: fp16 scatter to head-major Q [B,H,N,32] (out0)
// MODE 2: fp16 scatter: cols [0,192) -> K (out0), [192,384) -> V (out1)
template <typename AT, int MODE>
__global__ __launch_bounds__(256)
void gemm_wmma(const AT* __restrict__ A, const half_t* __restrict__ W,
               const float* __restrict__ bias, void* __restrict__ out0,
               void* __restrict__ out1, int M, int Ncol, int K) {
  __shared__ __attribute__((aligned(16))) half_t As[64 * 32];
  __shared__ __attribute__((aligned(16))) half_t Ws[64 * 32];

  const int tid  = threadIdx.x;
  const int lane = tid & 31;
  const int wave = tid >> 5;      // 0..7
  const int wm   = wave & 3;      // 16-row sub-tile
  const int wn   = wave >> 2;     // 0..1 -> 32-col sub-tile
  const int m0   = blockIdx.x * 64;
  const int n0   = blockIdx.y * 64;

  const int srow = tid >> 2;          // 0..63
  const int skg  = (tid & 3) * 8;     // 0,8,16,24

  v8f acc0 = {}; v8f acc1 = {};

  for (int k0 = 0; k0 < K; k0 += 32) {
    // stage A tile (64x32) with fp32->fp16 conversion, and W tile (64n x 32k)
    #pragma unroll
    for (int i = 0; i < 8; ++i)
      As[srow * 32 + skg + i] = (half_t)(float)A[(size_t)(m0 + srow) * K + k0 + skg + i];
    #pragma unroll
    for (int i = 0; i < 8; ++i)
      Ws[srow * 32 + skg + i] = W[(size_t)(n0 + srow) * K + k0 + skg + i];
    if (k0 + 32 < K)
      __builtin_prefetch(&A[(size_t)(m0 + srow) * K + k0 + 32 + skg], 0, 1);
    __syncthreads();

    v16h a  = load_afrag(As, 32, wm * 16, lane);
    v16h b0 = load_bfrag(Ws, 32, wn * 32, lane);
    v16h b1 = load_bfrag(Ws, 32, wn * 32 + 16, lane);
    acc0 = __builtin_amdgcn_wmma_f32_16x16x32_f16(false, a, false, b0, (short)0, acc0, false, false);
    acc1 = __builtin_amdgcn_wmma_f32_16x16x32_f16(false, a, false, b1, (short)0, acc1, false, false);
    __syncthreads();
  }

  // epilogue
  const int colA = n0 + wn * 32 + (lane & 15);
  const int colB = colA + 16;
  const float biasA = bias[colA];
  const float biasB = bias[colB];
  #pragma unroll
  for (int r = 0; r < 8; ++r) {
    const int row = m0 + wm * 16 + r + ((lane & 16) ? 8 : 0);
    const float vA = acc0[r] + biasA;
    const float vB = acc1[r] + biasB;
    if constexpr (MODE == 0) {
      float* o = (float*)out0;
      o[(size_t)row * Ncol + colA] = vA;
      o[(size_t)row * Ncol + colB] = vB;
    } else {
      const int b = row >> 9, n = row & 511;
      auto scatter = [&](int c, float v) {
        half_t* dst;
        int cc = c;
        if constexpr (MODE == 2) {
          dst = (cc < Cc) ? (half_t*)out0 : (half_t*)out1;
          if (cc >= Cc) cc -= Cc;
        } else {
          dst = (half_t*)out0;
        }
        const int h = cc >> 5, d = cc & 31;
        dst[((((size_t)b * Hc + h) * Nc + n) * Dc) + d] = (half_t)v;
      };
      scatter(colA, vA);
      scatter(colB, vB);
    }
  }
}

// ------------------------------------------------- fused attention kernel ----
// One block per (b,h). Qn pre-scaled+normalized, Kn normalized, all fp16
// [B,H,N,32].  Output: fp16 [B,N,C] (C index = h*32+d) for the final GEMM.
// Two-pass softmax: pass 1 finds exact row max with lane-partial maxima
// (one butterfly per M-tile); pass 2 recomputes S, applies exp with the fixed
// max (no rescale), accumulates lane-partial row sums (one butterfly at end).
__global__ __launch_bounds__(256)
void attn_kernel(const half_t* __restrict__ Qn, const half_t* __restrict__ Kn,
                 const half_t* __restrict__ Vh, half_t* __restrict__ AO) {
  __shared__ __attribute__((aligned(16))) half_t Ks[Nc * Dc];        // [key][d]
  __shared__ __attribute__((aligned(16))) half_t Vt[Dc * 520];       // [d][key], padded
  __shared__ __attribute__((aligned(16))) half_t Ps[8][16 * 32];     // per-wave P scratch

  const int bh   = blockIdx.x;          // b*6 + h
  const int tid  = threadIdx.x;
  const int lane = tid & 31;
  const int wave = tid >> 5;
  const size_t base = (size_t)bh * Nc * Dc;

  // load K row-major via 16B vectors
  {
    const h8* src = (const h8*)(Kn + base);
    h8* dst = (h8*)Ks;
    #pragma unroll
    for (int i = 0; i < (Nc * Dc / 8) / 256; ++i)
      dst[tid + i * 256] = src[tid + i * 256];
  }
  // load V transposed: Vt[d][key] = V[key][d]
  {
    #pragma unroll
    for (int i = 0; i < (Nc * Dc) / 256; ++i) {
      int e = tid + i * 256;
      int key = e >> 5, d = e & 31;
      Vt[d * 520 + key] = Vh[base + e];
    }
  }
  __syncthreads();

  half_t* Pw = Ps[wave];
  const int mbase = wave * 64;
  const int b = bh / Hc, h = bh % Hc;

  for (int mt = 0; mt < 4; ++mt) {
    const int mrow = mbase + mt * 16;
    const v16h aQ = load_afrag(Qn + base, Dc, mrow, lane);

    // ---- pass 1: exact row max, lane-partial (no cross-lane ops in loop)
    float mp[8];
    #pragma unroll
    for (int r = 0; r < 8; ++r) mp[r] = -1e30f;
    for (int j = 0; j < Nc; j += 32) {
      v16h bK0 = load_bfrag(Ks, Dc, j, lane);
      v16h bK1 = load_bfrag(Ks, Dc, j + 16, lane);
      v8f s0 = {}; v8f s1 = {};
      s0 = __builtin_amdgcn_wmma_f32_16x16x32_f16(false, aQ, false, bK0, (short)0, s0, false, false);
      s1 = __builtin_amdgcn_wmma_f32_16x16x32_f16(false, aQ, false, bK1, (short)0, s1, false, false);
      #pragma unroll
      for (int r = 0; r < 8; ++r) mp[r] = fmaxf(mp[r], fmaxf(s0[r], s1[r]));
    }
    #pragma unroll
    for (int r = 0; r < 8; ++r) mp[r] = redmax16(mp[r]);   // one butterfly / tile

    // ---- pass 2: P = exp(S - m), accumulate O = P*V and lane-partial sums
    float ls[8];
    #pragma unroll
    for (int r = 0; r < 8; ++r) ls[r] = 0.0f;
    v8f o0 = {}; v8f o1 = {};

    for (int j = 0; j < Nc; j += 32) {
      v16h bK0 = load_bfrag(Ks, Dc, j, lane);
      v16h bK1 = load_bfrag(Ks, Dc, j + 16, lane);
      v8f s0 = {}; v8f s1 = {};
      s0 = __builtin_amdgcn_wmma_f32_16x16x32_f16(false, aQ, false, bK0, (short)0, s0, false, false);
      s1 = __builtin_amdgcn_wmma_f32_16x16x32_f16(false, aQ, false, bK1, (short)0, s1, false, false);

      #pragma unroll
      for (int r = 0; r < 8; ++r) {
        const float p0 = __expf(s0[r] - mp[r]);
        const float p1 = __expf(s1[r] - mp[r]);
        ls[r] += p0 + p1;
        const int prow = r + ((lane & 16) ? 8 : 0);
        const int pcol = lane & 15;
        Pw[prow * 32 + pcol]      = (half_t)p0;
        Pw[prow * 32 + pcol + 16] = (half_t)p1;
      }

      // O += P (16x32) * V (32keys x 32d)  — per-wave LDS, in-wave ordering
      v16h aP  = load_afrag(Pw, 32, 0, lane);
      v16h bV0 = load_bfrag(Vt + j, 520, 0, lane);   // d = 0..15
      v16h bV1 = load_bfrag(Vt + j, 520, 16, lane);  // d = 16..31
      o0 = __builtin_amdgcn_wmma_f32_16x16x32_f16(false, aP, false, bV0, (short)0, o0, false, false);
      o1 = __builtin_amdgcn_wmma_f32_16x16x32_f16(false, aP, false, bV1, (short)0, o1, false, false);
    }

    #pragma unroll
    for (int r = 0; r < 8; ++r) ls[r] = redsum16(ls[r]);   // one butterfly / tile

    // write normalized O to AO [B,N,C], C = h*32 + d
    const int d0 = lane & 15, d1 = d0 + 16;
    #pragma unroll
    for (int r = 0; r < 8; ++r) {
      const int grow = mrow + r + ((lane & 16) ? 8 : 0);
      const float inv = 1.0f / fmaxf(ls[r], 1e-37f);
      half_t* dst = AO + ((size_t)(b * Nc + grow)) * Cc + h * Dc;
      dst[d0] = (half_t)(o0[r] * inv);
      dst[d1] = (half_t)(o1[r] * inv);
    }
  }
}

// ---------------------------------------------------------------- launch ----
extern "C" void kernel_launch(void* const* d_in, const int* in_sizes, int n_in,
                              void* d_out, int out_size, void* d_ws, size_t ws_size,
                              hipStream_t stream) {
  const float* x1     = (const float*)d_in[0];
  const float* x2     = (const float*)d_in[1];
  const float* q_w    = (const float*)d_in[2];
  const float* q_b    = (const float*)d_in[3];
  const float* kv_w   = (const float*)d_in[4];
  const float* v_b    = (const float*)d_in[5];
  const float* lscale = (const float*)d_in[6];
  const float* proj_w = (const float*)d_in[7];
  const float* proj_b = (const float*)d_in[8];

  char* ws = (char*)d_ws;
  size_t off = 0;
  auto carve = [&](size_t bytes) { void* p = ws + off; off = (off + bytes + 255) & ~(size_t)255; return p; };

  half_t* Wq     = (half_t*)carve((size_t)Cc * Cc * 2);
  half_t* Wkv    = (half_t*)carve((size_t)2 * Cc * Cc * 2);
  half_t* Wproj  = (half_t*)carve((size_t)Cc * Cc * 2);
  float*  kvbias = (float*) carve((size_t)2 * Cc * 4);
  half_t* Qr     = (half_t*)carve((size_t)Mrows * Cc * 2);  // [B,H,N,32]
  half_t* Kr     = (half_t*)carve((size_t)Mrows * Cc * 2);
  half_t* Vr     = (half_t*)carve((size_t)Mrows * Cc * 2);
  half_t* AO     = (half_t*)carve((size_t)Mrows * Cc * 2);  // [B,N,C]

  // 1. weight conversion + kv bias
  f2h_kernel<<<(Cc * Cc + 255) / 256, 256, 0, stream>>>(q_w, Wq, Cc * Cc);
  f2h_kernel<<<(2 * Cc * Cc + 255) / 256, 256, 0, stream>>>(kv_w, Wkv, 2 * Cc * Cc);
  f2h_kernel<<<(Cc * Cc + 255) / 256, 256, 0, stream>>>(proj_w, Wproj, Cc * Cc);
  build_kvbias<<<2, 192, 0, stream>>>(v_b, kvbias);

  // 2. projections (WMMA), scattered to head-major fp16
  gemm_wmma<float, 1><<<dim3(Mrows / 64, Cc / 64), 256, 0, stream>>>(
      x1, Wq, q_b, Qr, nullptr, Mrows, Cc, Cc);
  gemm_wmma<float, 2><<<dim3(Mrows / 64, 2 * Cc / 64), 256, 0, stream>>>(
      x2, Wkv, kvbias, Kr, Vr, Mrows, 2 * Cc, Cc);

  // 3. l2 norm (scale folded into Q)
  const int nrows = Bc * Hc * Nc;
  l2norm_rows<<<(nrows + 255) / 256, 256, 0, stream>>>(Qr, lscale, nrows);
  l2norm_rows<<<(nrows + 255) / 256, 256, 0, stream>>>(Kr, nullptr, nrows);

  // 4. fused attention, one block per (b,h)
  attn_kernel<<<Bc * Hc, 256, 0, stream>>>(Qr, Kr, Vr, AO);

  // 5. output projection -> fp32 d_out
  gemm_wmma<half_t, 0><<<dim3(Mrows / 64, Cc / 64), 256, 0, stream>>>(
      AO, Wproj, proj_b, (float*)d_out, nullptr, Mrows, Cc, Cc);
}
